// MolGNNLayers_34136400068697
// MI455X (gfx1250) — compile-verified, matrix operand order checked by reading, not run on hardware
//
#include <hip/hip_runtime.h>
#include <hip/hip_bf16.h>
#include <math.h>

typedef __attribute__((ext_vector_type(16))) _Float16 v16h;
typedef __attribute__((ext_vector_type(8)))  float    v8f;

#define D_MODEL 128
#define NHEAD   4
#define CH      32
#define NLAYER  4
#define NEG_SLOPE 0.2f
#define CDIV(a,b) (((a)+(b)-1)/(b))

// ---- order-preserving float <-> uint for atomicMax-based segment max ----
__device__ __forceinline__ unsigned fenc(float f) {
    unsigned u = __float_as_uint(f);
    return (u & 0x80000000u) ? ~u : (u | 0x80000000u);
}
__device__ __forceinline__ float fdec(unsigned u) {
    return __uint_as_float((u & 0x80000000u) ? (u & 0x7FFFFFFFu) : ~u);
}

// ---------------------------------------------------------------- utilities
__global__ void k_zero_f32(float* __restrict__ p, int n) {
    int t = blockIdx.x * blockDim.x + threadIdx.x;
    if (t < n) p[t] = 0.0f;
}

// Pack Wl/Wr (L x 128 x 128 f32) into f16 WMMA B-fragment layout:
// flat id = ((((l*2+m)*4+kb)*8+nt)*32+lane)*16+j
// element (K,N): K = kb*32 + (j<8 ? j : j+8) + (lane>=16 ? 8 : 0), N = nt*16 + (lane&15)
__global__ void k_pack_w(const float* __restrict__ Wl, const float* __restrict__ Wr,
                         _Float16* __restrict__ Wpk, int total) {
    int id = blockIdx.x * blockDim.x + threadIdx.x;
    if (id >= total) return;
    int j    = id & 15;
    int lane = (id >> 4) & 31;
    int nt   = (id >> 9) & 7;
    int kb   = (id >> 12) & 3;
    int m    = (id >> 14) & 1;
    int l    = id >> 15;
    int K = kb * 32 + (j < 8 ? j : j + 8) + ((lane >= 16) ? 8 : 0);
    int N = nt * 16 + (lane & 15);
    const float* W = (m ? Wr : Wl) + (size_t)l * D_MODEL * D_MODEL;
    Wpk[id] = (_Float16)W[(size_t)K * D_MODEL + N];
}

// Atom encoder: h[n,:] = sum_f atom_tables[f, x[n,f], :]
__global__ void k_atom_encode(const float* __restrict__ tabs, const int* __restrict__ x,
                              float* __restrict__ h, int n_nodes) {
    int t = blockIdx.x * blockDim.x + threadIdx.x;
    int n = t >> 5, q = t & 31;
    if (n >= n_nodes) return;
    float4 acc = make_float4(0.f, 0.f, 0.f, 0.f);
    #pragma unroll
    for (int f = 0; f < 9; ++f) {
        int idx = x[n * 9 + f];
        const float4* row = (const float4*)(tabs + (size_t)(f * 16 + idx) * D_MODEL);
        float4 v = row[q];
        acc.x += v.x; acc.y += v.y; acc.z += v.z; acc.w += v.w;
    }
    ((float4*)(h + (size_t)n * D_MODEL))[q] = acc;
}

// Bond encoder + per-dst sums/counts for self-loop 'mean' fill
__global__ void k_bond_scatter(const float* __restrict__ bond_table, const int* __restrict__ edge_attr,
                               const int* __restrict__ edge_index, float* __restrict__ e_full,
                               float* __restrict__ esum, float* __restrict__ ecnt, int E) {
    int e = blockIdx.x * blockDim.x + threadIdx.x;
    if (e >= E) return;
    int bt  = edge_attr[e];
    int dst = edge_index[E + e];
    #pragma unroll
    for (int j = 0; j < NHEAD; ++j) {
        float v = bond_table[bt * NHEAD + j];
        e_full[(size_t)e * NHEAD + j] = v;
        atomicAdd(esum + (size_t)dst * NHEAD + j, v);
    }
    atomicAdd(ecnt + dst, 1.0f);
}

__global__ void k_loop_attr(const float* __restrict__ esum, const float* __restrict__ ecnt,
                            float* __restrict__ e_full, int E, int n_nodes) {
    int t = blockIdx.x * blockDim.x + threadIdx.x;
    int n = t >> 2, j = t & 3;
    if (n >= n_nodes) return;
    float c = fmaxf(ecnt[n], 1.0f);
    e_full[((size_t)E + n) * NHEAD + j] = esum[(size_t)n * NHEAD + j] / c;
}

// GEMM: xl = h @ Wl + bl, xr = h @ Wr + br.  One wave = 16 rows x 128 cols,
// K=128 in 4 steps of 32 -> 64 v_wmma_f32_16x16x32_f16 per wave (A reused for L and R).
__global__ __launch_bounds__(256) void k_gemm_xlr(const float* __restrict__ h,
        const _Float16* __restrict__ Wpk_layer, const float* __restrict__ bl,
        const float* __restrict__ br, float* __restrict__ xl, float* __restrict__ xr,
        int n_nodes, int numTiles) {
    __shared__ v16h ldsW[2048];            // 64 KB: [m][kb][nt][lane] fragments
    {
        const uint4* src = (const uint4*)Wpk_layer;
        uint4* dstp = (uint4*)ldsW;
        for (int i = threadIdx.x; i < 4096; i += 256) dstp[i] = src[i];
    }
    __syncthreads();

    int wave = threadIdx.x >> 5, lane = threadIdx.x & 31;
    int tile = blockIdx.x * 8 + wave;
    if (tile >= numTiles) return;

    int row  = tile * 16 + (lane & 15);
    int rowc = (row < n_nodes) ? row : (n_nodes - 1);
    const float4* h4 = (const float4*)(h + (size_t)rowc * D_MODEL);
    int hiK = (lane >= 16) ? 8 : 0;

    v8f accL[8], accR[8];
    #pragma unroll
    for (int nt = 0; nt < 8; ++nt) {
        v8f z = {0.f, 0.f, 0.f, 0.f, 0.f, 0.f, 0.f, 0.f};
        accL[nt] = z; accR[nt] = z;
    }

    #pragma unroll
    for (int kb = 0; kb < 4; ++kb) {
        int k0 = kb * 32 + hiK;
        float4 p0 = h4[(k0 >> 2) + 0];
        float4 p1 = h4[(k0 >> 2) + 1];
        float4 p2 = h4[(k0 >> 2) + 4];
        float4 p3 = h4[(k0 >> 2) + 5];
        v16h a;
        a[0]=(_Float16)p0.x;  a[1]=(_Float16)p0.y;  a[2]=(_Float16)p0.z;  a[3]=(_Float16)p0.w;
        a[4]=(_Float16)p1.x;  a[5]=(_Float16)p1.y;  a[6]=(_Float16)p1.z;  a[7]=(_Float16)p1.w;
        a[8]=(_Float16)p2.x;  a[9]=(_Float16)p2.y;  a[10]=(_Float16)p2.z; a[11]=(_Float16)p2.w;
        a[12]=(_Float16)p3.x; a[13]=(_Float16)p3.y; a[14]=(_Float16)p3.z; a[15]=(_Float16)p3.w;
        #pragma unroll
        for (int nt = 0; nt < 8; ++nt) {
            v16h bL = ldsW[(kb * 8 + nt) * 32 + lane];
            accL[nt] = __builtin_amdgcn_wmma_f32_16x16x32_f16(
                false, a, false, bL, (short)0, accL[nt], false, false);
            v16h bR = ldsW[1024 + (kb * 8 + nt) * 32 + lane];
            accR[nt] = __builtin_amdgcn_wmma_f32_16x16x32_f16(
                false, a, false, bR, (short)0, accR[nt], false, false);
        }
    }

    int rbase = tile * 16 + ((lane >= 16) ? 8 : 0);
    #pragma unroll
    for (int nt = 0; nt < 8; ++nt) {
        int col = nt * 16 + (lane & 15);
        float bLv = bl[col], bRv = br[col];
        #pragma unroll
        for (int r = 0; r < 8; ++r) {
            int orow = rbase + r;
            if (orow < n_nodes) {
                xl[(size_t)orow * D_MODEL + col] = accL[nt][r] + bLv;
                xr[(size_t)orow * D_MODEL + col] = accR[nt][r] + bRv;
            }
        }
    }
}

// Pass 1: logits[e,h] = sum_c leaky(xl[src]+xr[dst]+ee) * att ; atomicMax per dst
__global__ __launch_bounds__(256) void k_attn_logits(const float* __restrict__ xl,
        const float* __restrict__ xr, const float* __restrict__ e_full,
        const float* __restrict__ Wel, const float* __restrict__ attl,
        const int* __restrict__ edge_index, float* __restrict__ logits,
        unsigned* __restrict__ mmax, int E, int Ef) {
    __shared__ float sWe[NHEAD * D_MODEL];
    __shared__ float sAtt[D_MODEL];
    for (int i = threadIdx.x; i < NHEAD * D_MODEL; i += 256) sWe[i] = Wel[i];
    for (int i = threadIdx.x; i < D_MODEL; i += 256) sAtt[i] = attl[i];
    __syncthreads();

    int t = blockIdx.x * blockDim.x + threadIdx.x;
    int e = t >> 2, hh = t & 3;
    if (e >= Ef) return;
    int s, d;
    if (e < E) { s = edge_index[e]; d = edge_index[E + e]; } else { s = d = e - E; }
    float ef0 = e_full[(size_t)e * 4 + 0], ef1 = e_full[(size_t)e * 4 + 1];
    float ef2 = e_full[(size_t)e * 4 + 2], ef3 = e_full[(size_t)e * 4 + 3];
    const float* xls = xl + (size_t)s * D_MODEL + hh * CH;
    const float* xrd = xr + (size_t)d * D_MODEL + hh * CH;
    float acc = 0.f;
    #pragma unroll
    for (int c = 0; c < CH; ++c) {
        float ee = ef0 * sWe[0 * D_MODEL + hh * CH + c] + ef1 * sWe[1 * D_MODEL + hh * CH + c]
                 + ef2 * sWe[2 * D_MODEL + hh * CH + c] + ef3 * sWe[3 * D_MODEL + hh * CH + c];
        float g = xls[c] + xrd[c] + ee;
        g = (g > 0.f) ? g : NEG_SLOPE * g;
        acc += g * sAtt[hh * CH + c];
    }
    logits[(size_t)e * 4 + hh] = acc;
    atomicMax(mmax + (size_t)d * 4 + hh, fenc(acc));
}

// Pass 2: w = exp(logit - max); z[dst] += w
__global__ void k_attn_expsum(const int* __restrict__ edge_index, float* __restrict__ logits,
                              const unsigned* __restrict__ mmax, float* __restrict__ zsum,
                              int E, int Ef) {
    int t = blockIdx.x * blockDim.x + threadIdx.x;
    int e = t >> 2, hh = t & 3;
    if (e >= Ef) return;
    int d = (e < E) ? edge_index[E + e] : (e - E);
    float m = fdec(mmax[(size_t)d * 4 + hh]);
    float w = __expf(logits[(size_t)e * 4 + hh] - m);
    logits[(size_t)e * 4 + hh] = w;
    atomicAdd(zsum + (size_t)d * 4 + hh, w);
}

// Pass 3: h[dst,:] += (w / (z+eps)) * xl[src,:]
__global__ void k_aggregate(const int* __restrict__ edge_index, const float* __restrict__ xl,
                            const float* __restrict__ logits, const float* __restrict__ zsum,
                            float* __restrict__ hout, int E, int Ef) {
    int t = blockIdx.x * blockDim.x + threadIdx.x;
    int e = t >> 7, dd = t & 127;
    if (e >= Ef) return;
    int s, d;
    if (e < E) { s = edge_index[e]; d = edge_index[E + e]; } else { s = d = e - E; }
    int hh = dd >> 5;
    float w = logits[(size_t)e * 4 + hh];
    float z = zsum[(size_t)d * 4 + hh];
    float alpha = w / (z + 1e-16f);
    atomicAdd(hout + (size_t)d * D_MODEL + dd, alpha * xl[(size_t)s * D_MODEL + dd]);
}

__global__ void k_bias_act(float* __restrict__ h, const float* __restrict__ cbl,
                           int n, int do_relu) {
    int t = blockIdx.x * blockDim.x + threadIdx.x;
    if (t >= n) return;
    float v = h[t] + cbl[t & 127];
    h[t] = do_relu ? fmaxf(v, 0.f) : v;
}

__global__ void k_pool(const float* __restrict__ h, const int* __restrict__ batch_ids,
                       float* __restrict__ pooled, int total) {
    int t = blockIdx.x * blockDim.x + threadIdx.x;
    if (t >= total) return;
    int n = t >> 7, d = t & 127;
    atomicAdd(pooled + (size_t)batch_ids[n] * D_MODEL + d, h[t]);
}

__global__ void k_head(const float* __restrict__ pooled, const float* __restrict__ Wout,
                       const float* __restrict__ bout, float* __restrict__ out, int G, int T) {
    int g = blockIdx.x * blockDim.x + threadIdx.x;
    if (g >= G) return;
    const float* p = pooled + (size_t)g * D_MODEL;
    for (int tk = 0; tk < T; ++tk) {
        float acc = bout[tk];
        #pragma unroll 4
        for (int d = 0; d < D_MODEL; ++d) acc += p[d] * Wout[d * T + tk];
        out[g * T + tk] = acc;
    }
}

extern "C" void kernel_launch(void* const* d_in, const int* in_sizes, int n_in,
                              void* d_out, int out_size, void* d_ws, size_t ws_size,
                              hipStream_t stream) {
    (void)n_in; (void)ws_size;
    const float* atom_tables = (const float*)d_in[0];
    const float* bond_table  = (const float*)d_in[1];
    const float* Wl   = (const float*)d_in[2];
    const float* bl   = (const float*)d_in[3];
    const float* Wr   = (const float*)d_in[4];
    const float* br   = (const float*)d_in[5];
    const float* We   = (const float*)d_in[6];
    const float* att  = (const float*)d_in[7];
    const float* cb   = (const float*)d_in[8];
    const float* Wout = (const float*)d_in[9];
    const float* bout = (const float*)d_in[10];
    const int* x          = (const int*)d_in[11];
    const int* edge_index = (const int*)d_in[12];
    const int* edge_attr  = (const int*)d_in[13];
    const int* batch_ids  = (const int*)d_in[14];

    const int N  = in_sizes[11] / 9;
    const int E  = in_sizes[12] / 2;
    const int T  = in_sizes[9] / D_MODEL;   // NTASK
    const int G  = out_size / T;
    const int Ef = E + N;

    char* ws = (char*)d_ws;
    size_t off = 0;
    auto alloc = [&](size_t bytes) -> char* {
        char* p = ws + off;
        off += (bytes + 255) & ~(size_t)255;
        return p;
    };
    float*     hbuf   = (float*)alloc((size_t)N * D_MODEL * 4);
    float*     xl     = (float*)alloc((size_t)N * D_MODEL * 4);
    float*     xr     = (float*)alloc((size_t)N * D_MODEL * 4);
    float*     e_full = (float*)alloc((size_t)Ef * NHEAD * 4);
    float*     logits = (float*)alloc((size_t)Ef * NHEAD * 4);
    unsigned*  mmax   = (unsigned*)alloc((size_t)N * NHEAD * 4);
    float*     zsum   = (float*)alloc((size_t)N * NHEAD * 4);
    float*     esum   = (float*)alloc((size_t)N * NHEAD * 4);
    float*     ecnt   = (float*)alloc((size_t)N * 4);
    _Float16*  Wpk    = (_Float16*)alloc((size_t)NLAYER * 2 * D_MODEL * D_MODEL * 2);
    float*     pooled = (float*)alloc((size_t)G * D_MODEL * 4);

    // Setup: pack weights, encode atoms/bonds, self-loop mean fill
    {
        int total = NLAYER * 2 * D_MODEL * D_MODEL;
        k_pack_w<<<CDIV(total, 256), 256, 0, stream>>>(Wl, Wr, Wpk, total);
    }
    k_atom_encode<<<CDIV(N * 32, 256), 256, 0, stream>>>(atom_tables, x, hbuf, N);
    k_zero_f32<<<CDIV(N * NHEAD, 256), 256, 0, stream>>>(esum, N * NHEAD);
    k_zero_f32<<<CDIV(N, 256), 256, 0, stream>>>(ecnt, N);
    k_bond_scatter<<<CDIV(E, 256), 256, 0, stream>>>(bond_table, edge_attr, edge_index,
                                                     e_full, esum, ecnt, E);
    k_loop_attr<<<CDIV(N * NHEAD, 256), 256, 0, stream>>>(esum, ecnt, e_full, E, N);

    const int numTiles = CDIV(N, 16);
    for (int l = 0; l < NLAYER; ++l) {
        k_gemm_xlr<<<CDIV(numTiles, 8), 256, 0, stream>>>(
            hbuf, Wpk + (size_t)l * 2 * D_MODEL * D_MODEL,
            bl + l * D_MODEL, br + l * D_MODEL, xl, xr, N, numTiles);
        k_zero_f32<<<CDIV(N * NHEAD, 256), 256, 0, stream>>>((float*)mmax, N * NHEAD);
        k_zero_f32<<<CDIV(N * NHEAD, 256), 256, 0, stream>>>(zsum, N * NHEAD);
        k_attn_logits<<<CDIV(Ef * NHEAD, 256), 256, 0, stream>>>(
            xl, xr, e_full, We + (size_t)l * NHEAD * D_MODEL, att + (size_t)l * D_MODEL,
            edge_index, logits, mmax, E, Ef);
        k_attn_expsum<<<CDIV(Ef * NHEAD, 256), 256, 0, stream>>>(
            edge_index, logits, mmax, zsum, E, Ef);
        k_zero_f32<<<CDIV(N * D_MODEL, 256), 256, 0, stream>>>(hbuf, N * D_MODEL);
        k_aggregate<<<CDIV(Ef * D_MODEL, 256), 256, 0, stream>>>(
            edge_index, xl, logits, zsum, hbuf, E, Ef);
        k_bias_act<<<CDIV(N * D_MODEL, 256), 256, 0, stream>>>(
            hbuf, cb + l * D_MODEL, N * D_MODEL, (l < NLAYER - 1) ? 1 : 0);
    }

    k_zero_f32<<<CDIV(G * D_MODEL, 256), 256, 0, stream>>>(pooled, G * D_MODEL);
    k_pool<<<CDIV(N * D_MODEL, 256), 256, 0, stream>>>(hbuf, batch_ids, pooled, N * D_MODEL);
    k_head<<<CDIV(G, 256), 256, 0, stream>>>(pooled, Wout, bout, (float*)d_out, G, T);
}
